// OptimizedLSTM_76115410420291
// MI455X (gfx1250) — matrix operand order repaired and checked by reading, not compiled, not run on hardware
//
#include <hip/hip_runtime.h>
#include <hip/hip_bf16.h>

// ---------------------------------------------------------------------------
// CDNA5 WMMA types
// ---------------------------------------------------------------------------
typedef __attribute__((ext_vector_type(16))) __bf16 v16bf;
typedef __attribute__((ext_vector_type(8)))  __bf16 v8bf;
typedef __attribute__((ext_vector_type(4)))  __bf16 v4bf;
typedef __attribute__((ext_vector_type(8)))  float  v8f;

union V16U { v16bf v; v8bf h[2]; };

// A-matrix 16x32 bf16 fragment: lane<16 holds row M=lane,
// K = 0..15; lane>=16 holds row M=lane-16, K = 16..31.
__device__ __forceinline__ v16bf load_a_frag(const __bf16* base, int stride,
                                             int m0, int k0, int lane) {
    int r = lane & 15, half = lane >> 4;
    const __bf16* p = base + (size_t)(m0 + r) * stride + k0 + half * 16;
    V16U u;
    u.h[0] = *(const v8bf*)(p);
    u.h[1] = *(const v8bf*)(p + 8);
    return u.v;
}

// B-matrix 32x16 bf16 fragment: B[k][n] = W[n][k] (W row-major), lane<16
// holds column N=lane, K=k0..k0+15 (contiguous 32B of W row n0+lane);
// lane>=16 holds column N=lane-16, K=k0+16..k0+31.
__device__ __forceinline__ v16bf load_b_frag(const __bf16* base, int stride,
                                             int n0, int k0, int lane) {
    int r = lane & 15, half = lane >> 4;
    const __bf16* p = base + (size_t)(n0 + r) * stride + k0 + half * 16;
    return *(const v16bf*)(p);
}

__device__ __forceinline__ v8f wmma_bf16(v16bf a, v16bf b, v8f c) {
    // (neg_a, A, neg_b, B, c_mod, C, reuse_a, reuse_b)
    return __builtin_amdgcn_wmma_f32_16x16x32_bf16(false, a, false, b,
                                                   (short)0, c, false, false);
}

__device__ __forceinline__ float sigmoidf_fast(float x) {
    return 1.0f / (1.0f + __expf(-x));
}

#define VOCAB  100000
#define INPUT  256
#define HIDDEN 512
#define GATES  2048      // 4*HIDDEN
#define BATCH  64
#define SEQ    512
#define NROWS  (BATCH * SEQ)   // 32768

// ---------------------------------------------------------------------------
// Phase 0a: f32 -> bf16 weight conversion
// ---------------------------------------------------------------------------
__global__ void convert_weights_kernel(const float* __restrict__ Wih,
                                       const float* __restrict__ Whh,
                                       __bf16* __restrict__ wih_bf,
                                       __bf16* __restrict__ whh_bf) {
    const int n1 = GATES * INPUT;    // 524288
    const int n2 = GATES * HIDDEN;   // 1048576
    for (int i = blockIdx.x * blockDim.x + threadIdx.x; i < n1 + n2;
         i += gridDim.x * blockDim.x) {
        if (i < n1) wih_bf[i] = (__bf16)Wih[i];
        else        whh_bf[i - n1] = (__bf16)Whh[i - n1];
    }
}

// ---------------------------------------------------------------------------
// Phase 0b: embedding gather (+ padding_idx=0 mask) -> bf16 A matrix [32768,256]
// ---------------------------------------------------------------------------
__global__ void gather_embed_kernel(const int* __restrict__ x,
                                    const float* __restrict__ emb,
                                    __bf16* __restrict__ A) {
    int row = blockIdx.x;            // b*SEQ + s
    int idx = x[row];
    float v = (idx == 0) ? 0.0f : emb[(size_t)idx * INPUT + threadIdx.x];
    A[(size_t)row * INPUT + threadIdx.x] = (__bf16)v;
}

// ---------------------------------------------------------------------------
// Phase 1: x_proj = Aemb @ W_ih^T  (M=32768, N=2048, K=256), bf16 out, no bias.
// Output is GATE-INTERLEAVED: xproj[row][j*4 + gate]  (gate = col>>9, j = col&511)
// so the recurrence reads one 8-byte v4bf per LSTM cell instead of 4 scattered
// 2-byte loads. Block = 256 threads (8 waves); tile 128(M) x 64(N); wave 16x64.
// ---------------------------------------------------------------------------
__global__ void xproj_gemm_kernel(const __bf16* __restrict__ Aemb,
                                  const __bf16* __restrict__ Wih,
                                  __bf16* __restrict__ xproj) {
    const int wid  = threadIdx.x >> 5;
    const int lane = threadIdx.x & 31;
    const int nb = blockIdx.x & 31;          // 32 N blocks of 64
    const int mb = blockIdx.x >> 5;          // 256 M blocks of 128
    const int m0 = mb * 128 + wid * 16;
    const int n0 = nb * 64;

    v8f acc[4] = {};
    for (int k0 = 0; k0 < INPUT; k0 += 32) {
        v16bf a = load_a_frag(Aemb, INPUT, m0, k0, lane);
#pragma unroll
        for (int ns = 0; ns < 4; ++ns) {
            v16bf bf = load_b_frag(Wih, INPUT, n0 + ns * 16, k0, lane);
            acc[ns] = wmma_bf16(a, bf, acc[ns]);
        }
    }
    const int r = lane & 15, half = lane >> 4;
#pragma unroll
    for (int ns = 0; ns < 4; ++ns) {
        int col  = n0 + ns * 16 + r;
        int gate = col >> 9;                 // 0..3 (i,f,g,o)
        int j    = col & (HIDDEN - 1);       // hidden index
#pragma unroll
        for (int jj = 0; jj < 8; ++jj) {
            int row = m0 + jj + half * 8;
            xproj[(size_t)row * GATES + j * 4 + gate] = (__bf16)acc[ns][jj];
        }
    }
}

// ---------------------------------------------------------------------------
// Phase 2: persistent LSTM recurrence. Grid = 16 WGs x 256 threads.
// Each WG owns 32 hidden columns; W_hh slice (4 gates x 32 rows x 512 K,
// 128 KB bf16) staged in LDS once, reused for all 512 timesteps.
// Wave w: m = w>>1 (batch 16-row strip), n = w&1 (hidden 16-col strip);
// computes all 4 gate tiles for its 16x16 cell block -> in-register LSTM cell.
// c-state lives in registers (static ownership); h double-buffered bf16 in ws.
// Grid-wide sync per step via monotonic atomic counter (16 WGs co-resident).
// ---------------------------------------------------------------------------
__global__ void lstm_recurrence_kernel(const __bf16* __restrict__ xproj,
                                       const __bf16* __restrict__ Whh,
                                       const float*  __restrict__ b_ih,
                                       const float*  __restrict__ b_hh,
                                       __bf16* __restrict__ hbuf,   // 2 x [64x512]
                                       float*  __restrict__ hout,   // d_out h region
                                       float*  __restrict__ cout,   // d_out c region
                                       unsigned* __restrict__ counter) {
    extern __shared__ __bf16 lwhh[];   // [128][512] bf16 = 128 KB
    const int wg   = blockIdx.x;       // 0..15
    const int tid  = threadIdx.x;
    const int wid  = tid >> 5;
    const int lane = tid & 31;
    const int r    = lane & 15;
    const int half = lane >> 4;
    const unsigned nwg = gridDim.x;

    // Stage W_hh slice: for gate g, rows g*512 + wg*32 + rr -> LDS row g*32+rr.
    for (int ch = tid; ch < 128 * 32; ch += 256) {       // 32B chunks
        int row = ch >> 5, chunk = ch & 31;
        int g = row >> 5, rr = row & 31;
        const v8bf* src = (const v8bf*)(Whh + (size_t)(g * HIDDEN + wg * 32 + rr) * HIDDEN
                                        + chunk * 16);
        v8bf* dst = (v8bf*)(lwhh + (size_t)row * HIDDEN + chunk * 16);
        dst[0] = src[0];
        dst[1] = src[1];
    }
    __syncthreads();

    const int m = wid >> 1;            // batch strip 0..3
    const int n = wid & 1;             // hidden strip 0..1
    const int jcol = wg * 32 + n * 16 + r;   // hidden column owned by this lane

    // Hoisted per-lane constants (fixed across all timesteps).
    const float bias_i = b_ih[jcol]              + b_hh[jcol];
    const float bias_f = b_ih[HIDDEN + jcol]     + b_hh[HIDDEN + jcol];
    const float bias_g = b_ih[2 * HIDDEN + jcol] + b_hh[2 * HIDDEN + jcol];
    const float bias_o = b_ih[3 * HIDDEN + jcol] + b_hh[3 * HIDDEN + jcol];
    const __bf16* xpbase = xproj + (size_t)jcol * 4;   // + (bi*SEQ + t)*GATES

    float creg[8];                     // register-resident cell state
#pragma unroll
    for (int j = 0; j < 8; ++j) creg[j] = 0.0f;

    for (int t = 0; t < SEQ; ++t) {
        const __bf16* hin  = hbuf + (size_t)(t & 1) * (BATCH * HIDDEN);
        __bf16* hnext      = hbuf + (size_t)((t + 1) & 1) * (BATCH * HIDDEN);

        v8f acc[4] = {};   // i, f, g, o tiles
        for (int k0 = 0; k0 < HIDDEN; k0 += 32) {
            v16bf a = load_a_frag(hin, HIDDEN, m * 16, k0, lane);
#pragma unroll
            for (int g = 0; g < 4; ++g) {
                v16bf bf = load_b_frag(lwhh, HIDDEN, g * 32 + n * 16, k0, lane);
                acc[g] = wmma_bf16(a, bf, acc[g]);
            }
        }

#pragma unroll
        for (int j = 0; j < 8; ++j) {
            int bi = m * 16 + j + half * 8;                 // batch row
            const __bf16* xp = xpbase + ((size_t)bi * SEQ + t) * GATES;
            v4bf xv = *(const v4bf*)xp;                     // i,f,g,o interleaved
            float gi = acc[0][j] + (float)xv[0] + bias_i;
            float gf = acc[1][j] + (float)xv[1] + bias_f;
            float gg = acc[2][j] + (float)xv[2] + bias_g;
            float go = acc[3][j] + (float)xv[3] + bias_o;
            float iv = sigmoidf_fast(gi);
            float fv = sigmoidf_fast(gf);
            float gv = tanhf(gg);
            float ov = sigmoidf_fast(go);
            float cv = fv * creg[j] + iv * gv;
            creg[j] = cv;
            float hv = ov * tanhf(cv);
            size_t hidx = (size_t)bi * HIDDEN + jcol;
            hnext[hidx] = (__bf16)hv;
            if (t == SEQ - 1) {
                hout[hidx] = hv;
                cout[hidx] = cv;
            }
            // Hide next step's fresh-from-HBM x_proj latency behind this step.
            if (t < SEQ - 1) __builtin_prefetch(xp + GATES, 0, 0);
        }

        // ---- grid-wide barrier: h(t+1) fully written before anyone reads it
        __threadfence();
        __syncthreads();
        if (tid == 0) {
            unsigned target = (unsigned)(t + 1) * nwg;
            __hip_atomic_fetch_add(counter, 1u, __ATOMIC_RELEASE,
                                   __HIP_MEMORY_SCOPE_AGENT);
            while (__hip_atomic_load(counter, __ATOMIC_ACQUIRE,
                                     __HIP_MEMORY_SCOPE_AGENT) < target) {
                __builtin_amdgcn_s_sleep(1);
            }
        }
        __syncthreads();
        __threadfence();
    }
}

// ---------------------------------------------------------------------------
// Phase 3: out = h_T @ W_out^T + b_out   (64x512 x 512x5 -> tiny, scalar f32)
// ---------------------------------------------------------------------------
__global__ void out_head_kernel(const float* __restrict__ h,
                                const float* __restrict__ Wout,
                                const float* __restrict__ bout,
                                float* __restrict__ out) {
    int tid = blockIdx.x * blockDim.x + threadIdx.x;
    if (tid >= BATCH * 5) return;
    int b = tid / 5, o = tid % 5;
    float s = bout[o];
    const float* hr = h + (size_t)b * HIDDEN;
    const float* wr = Wout + (size_t)o * HIDDEN;
    for (int k = 0; k < HIDDEN; ++k) s += hr[k] * wr[k];
    out[tid] = s;
}

// ---------------------------------------------------------------------------
// Launch
// ---------------------------------------------------------------------------
extern "C" void kernel_launch(void* const* d_in, const int* in_sizes, int n_in,
                              void* d_out, int out_size, void* d_ws, size_t ws_size,
                              hipStream_t stream) {
    const int*   x         = (const int*)  d_in[0];
    const float* embedding = (const float*)d_in[1];
    const float* W_ih      = (const float*)d_in[2];
    const float* b_ih      = (const float*)d_in[3];
    const float* W_hh      = (const float*)d_in[4];
    const float* b_hh      = (const float*)d_in[5];
    const float* W_out     = (const float*)d_in[6];
    const float* b_out     = (const float*)d_in[7];

    // Workspace layout (all offsets 256B-aligned)
    char* ws = (char*)d_ws;
    size_t off = 0;
    unsigned* counter = (unsigned*)(ws + off);            off += 256;
    __bf16* wih_bf = (__bf16*)(ws + off);                 off += (size_t)GATES * INPUT * 2;   // 1 MB
    __bf16* whh_bf = (__bf16*)(ws + off);                 off += (size_t)GATES * HIDDEN * 2;  // 2 MB
    __bf16* hbuf   = (__bf16*)(ws + off);                 off += (size_t)2 * BATCH * HIDDEN * 2; // 128 KB
    __bf16* Aemb   = (__bf16*)(ws + off);                 off += (size_t)NROWS * INPUT * 2;   // 16 MB
    __bf16* xproj  = (__bf16*)(ws + off);                 off += (size_t)NROWS * GATES * 2;   // 128 MB

    float* out   = (float*)d_out;
    float* hout  = out + BATCH * 5;                 // [64,512] final h
    float* cout  = hout + BATCH * HIDDEN;           // [64,512] final c

    // Deterministic init (graph-capture safe); every d_out element is fully
    // overwritten each call, so no d_out memset is needed.
    hipMemsetAsync(counter, 0, 256, stream);
    hipMemsetAsync(hbuf, 0, (size_t)2 * BATCH * HIDDEN * 2, stream);

    convert_weights_kernel<<<512, 256, 0, stream>>>(W_ih, W_hh, wih_bf, whh_bf);
    gather_embed_kernel<<<NROWS, INPUT, 0, stream>>>(x, embedding, Aemb);
    xproj_gemm_kernel<<<256 * 32, 256, 0, stream>>>(Aemb, wih_bf, xproj);
    lstm_recurrence_kernel<<<16, 256, 128 * 1024, stream>>>(
        xproj, whh_bf, b_ih, b_hh, hbuf, hout, cout, counter);
    out_head_kernel<<<2, 160, 0, stream>>>(hout, W_out, b_out, out);
}
